// MotifGIN_39032662786179
// MI455X (gfx1250) — matrix operand order repaired and checked by reading, not compiled, not run on hardware
//
#include <hip/hip_runtime.h>

// ---------------------------------------------------------------------------
// MotifGIN on MI455X (gfx1250).
// Memory-bound (~6-8 GB traffic vs ~126 GFLOP) -> fp32 in HBM, bf16 only in
// LDS/VGPRs, GEMMs on v_wmma_f32_16x16x32_bf16 with fp32 accumulation.
// Each block: 64-row stripe, 8 waves x 8 col-tiles x 4 row-tiles.
// LDS: transposed weights [128][Kpad] (<=73KB) + A stripe [64][Kpad] (<=36KB)
// = <=110KB -> 2 workgroups per 320KB WGP. Single barrier, then a pure
// ds_load_b128 + v_wmma loop. Message buffers (154MB) live in the 192MB L2.
// ---------------------------------------------------------------------------

typedef __bf16 v16bf  __attribute__((ext_vector_type(16)));
typedef __bf16 bf16x8 __attribute__((ext_vector_type(8)));
typedef __bf16 bf16x4 __attribute__((ext_vector_type(4)));
typedef float  v8f    __attribute__((ext_vector_type(8)));

#define N_ATOMS   300000
#define N_MOTIFS  100000
#define N_GRAPHS  1000
#define E_ATOM    600000
#define E_MOTIF   400000
#define HID       128

// ---------------------------------------------------------------------------
// C[M,128] = act( (A (+Aadd)) [M,K] @ W[K,128] + bias )
// ---------------------------------------------------------------------------
__global__ __launch_bounds__(256)
void gemm_bias_wmma(const float* __restrict__ A, int lda, int aoff,
                    const float* __restrict__ Aadd,     // optional, same lda/aoff
                    const float* __restrict__ W,        // [K,128] row-major
                    const float* __restrict__ bias,     // [128]
                    float* __restrict__ C, int ldc, int coff,
                    int M, int K, int relu)
{
    extern __shared__ __bf16 smem[];
    const int Kpad = (K + 31) & ~31;
    __bf16* sW = smem;                    // [128][Kpad] (transposed weights)
    __bf16* sA = smem + 128 * Kpad;       // [64][Kpad]

    const int tid  = threadIdx.x;
    const int wave = tid >> 5;
    const int lane = tid & 31;
    const int rowBase = blockIdx.x * 64;

    // ---- stage weights to LDS, transposed [col][k], zero-padded in K ----
    for (int i = tid; i < 128 * Kpad; i += 256) {
        int c = i / Kpad, k = i - c * Kpad;
        sW[i] = (k < K) ? (__bf16)W[(long)k * 128 + c] : (__bf16)0.0f;
    }

    // ---- zero the pad columns of sA (disjoint from data fill: no race) ----
    if (Kpad != K) {
        int padw = Kpad - K;
        for (int i = tid; i < 64 * padw; i += 256) {
            int r = i / padw, k = K + (i - (i / padw) * padw);
            sA[r * Kpad + k] = (__bf16)0.0f;
        }
    }

    // ---- stage 64-row A stripe (fp32 -> bf16), fused optional add ----
    const bool fast = (lda == K) && (aoff == 0) && ((K & 3) == 0)
                      && (rowBase + 64 <= M);
    if (fast) {
        const float4* Ap  = (const float4*)(A + (long)rowBase * K);
        const float4* Ap2 = Aadd ? (const float4*)(Aadd + (long)rowBase * K)
                                 : nullptr;
        const int nvec = 16 * K;                    // 64*K/4 float4s
        for (int i = tid; i < nvec; i += 256) {
            float4 v = Ap[i];                       // global_load_b128
            __builtin_prefetch(&Ap[i + 256], 0, 0); // global_prefetch_b8
            if (Ap2) {
                float4 u = Ap2[i];
                v.x += u.x; v.y += u.y; v.z += u.z; v.w += u.w;
            }
            int f = i << 2;
            int r = f / K, k = f - r * K;           // k multiple of 4
            bf16x4 p;
            p[0] = (__bf16)v.x; p[1] = (__bf16)v.y;
            p[2] = (__bf16)v.z; p[3] = (__bf16)v.w;
            *(bf16x4*)(sA + r * Kpad + k) = p;      // ds_store_b64
        }
    } else {
        // tail block / strided slice: guarded scalar path
        for (int i = tid; i < 64 * Kpad; i += 256) {
            int r = i / Kpad, k = i - r * Kpad;
            int row = rowBase + r;
            float v = 0.0f;
            if (row < M && k < K) {
                long idx = (long)row * lda + aoff + k;
                v = A[idx];
                if (Aadd) v += Aadd[idx];
            }
            sA[r * Kpad + k] = (__bf16)v;
        }
    }
    __syncthreads();

    // ---- WMMA main loop: no barriers, B fragment reused by 4 row-tiles ----
    v8f acc[4] = {{}, {}, {}, {}};
    const int colT = wave * 16 + (lane & 15);       // absolute output col
    const int rA   = lane & 15;
    const int kbA  = (lane < 16) ? 0 : 8;           // A frag K sub-base
    const int kbB  = (lane < 16) ? 0 : 8;           // B frag K sub-base

    for (int k0 = 0; k0 < K; k0 += 32) {
        const __bf16* bp = sW + (long)colT * Kpad + k0 + kbB;
        bf16x8 blo = *(const bf16x8*)bp;            // ds_load_b128
        bf16x8 bhi = *(const bf16x8*)(bp + 16);
        v16bf bv;
#pragma unroll
        for (int i = 0; i < 8; ++i) { bv[i] = blo[i]; bv[i + 8] = bhi[i]; }

#pragma unroll
        for (int mt = 0; mt < 4; ++mt) {
            const __bf16* ap = sA + (mt * 16 + rA) * Kpad + k0 + kbA;
            bf16x8 alo = *(const bf16x8*)ap;        // K {0..7 | 8..15}
            bf16x8 ahi = *(const bf16x8*)(ap + 16); // K {16..23 | 24..31}
            v16bf av;
#pragma unroll
            for (int i = 0; i < 8; ++i) { av[i] = alo[i]; av[i + 8] = ahi[i]; }
            acc[mt] = __builtin_amdgcn_wmma_f32_16x16x32_bf16(
                          false, av, false, bv, (short)0, acc[mt], false, false);
        }
    }

    // ---- writeback: lane<16 -> rows 0..7, lane>=16 -> rows 8..15 ----
    float b  = bias[colT];
    int   mB = (lane < 16) ? 0 : 8;
#pragma unroll
    for (int mt = 0; mt < 4; ++mt) {
#pragma unroll
        for (int rr = 0; rr < 8; ++rr) {
            int row = rowBase + mt * 16 + mB + rr;
            if (row < M) {
                float v = acc[mt][rr] + b;
                if (relu) v = v > 0.0f ? v : 0.0f;
                C[(long)row * ldc + coff + colT] = v;
            }
        }
    }
}

// ---------------------------------------------------------------------------
// GINE message scatter: msg[dst[e]] += relu(x[src[e]] + eh[e])
// ---------------------------------------------------------------------------
__global__ void gine_scatter(const float* __restrict__ x,
                             const float* __restrict__ eh,
                             const int* __restrict__ src,
                             const int* __restrict__ dst,
                             float* __restrict__ msg)
{
    int e = blockIdx.x, f = threadIdx.x;
    int s = src[e], d = dst[e];
    float v = x[(long)s * HID + f] + eh[(long)e * HID + f];
    v = v > 0.0f ? v : 0.0f;
    atomicAdd(&msg[(long)d * HID + f], v);
}

// Pool atoms -> motifs: hm[mrow[a]] += x[arow[a]]
__global__ void pool_scatter(const float* __restrict__ x,
                             const int* __restrict__ mrow,
                             const int* __restrict__ arow,
                             float* __restrict__ hm)
{
    int a = blockIdx.x, f = threadIdx.x;
    atomicAdd(&hm[(long)mrow[a] * HID + f], x[(long)arow[a] * HID + f]);
}

// cat160[m] = [type_emb[motif_type[m]] (32) | h_motif[m] (128)]
__global__ void build_cat160(const float* __restrict__ type_emb,
                             const int* __restrict__ motif_type,
                             const float* __restrict__ hm,
                             float* __restrict__ cat)
{
    int m = blockIdx.x, t = threadIdx.x;      // blockDim = 160
    float v;
    if (t < 32) v = type_emb[motif_type[m] * 32 + t];
    else        v = hm[(long)m * HID + (t - 32)];
    cat[(long)m * 160 + t] = v;
}

// motif edge prep: cat288[:,0:32]=couple, cat288[:,160:288]=node_emb
// (cols 32:160 filled by the edge_emb GEMM)
__global__ void motif_edge_prep(const float* __restrict__ attr,   // [E,20]
                                const int* __restrict__ mei,      // [2,E]
                                const int* __restrict__ motif_batch,
                                const int* __restrict__ atom_ptr,
                                const float* __restrict__ type_emb,
                                const float* __restrict__ x_atom,
                                float* __restrict__ cat)          // [E,288]
{
    int e = blockIdx.x, t = threadIdx.x;      // blockDim = 128
    const float* a = attr + (long)e * 20;
    int msrc = mei[e];
    int off  = atom_ptr[motif_batch[msrc]];
    int t0   = (int)a[0], t1 = (int)a[1];
    int gdst = (int)a[18] + off;              // local_dst at col -2
    int gsrc = (int)a[19] + off;              // local_src at col -1
    float* row = cat + (long)e * 288;
    if (t < 32) row[t] = type_emb[t0 * 32 + t] + type_emb[t1 * 32 + t];
    row[160 + t] = x_atom[(long)gsrc * HID + t] + x_atom[(long)gdst * HID + t];
}

// Graph pool: ml[batch[m]] += h[m]
__global__ void graph_pool(const float* __restrict__ h,
                           const int* __restrict__ batch,
                           float* __restrict__ ml)
{
    int m = blockIdx.x, f = threadIdx.x;
    atomicAdd(&ml[(long)batch[m] * HID + f], h[(long)m * HID + f]);
}

// ---------------------------------------------------------------------------
static inline void run_gemm(const float* A, int lda, int aoff, const float* Aadd,
                            const float* W, const float* bias,
                            float* C, int ldc, int coff,
                            int M, int K, int relu, hipStream_t s)
{
    int Kpad = (K + 31) & ~31;
    size_t sh = (size_t)(128 * Kpad + 64 * Kpad) * 2;  // bf16 bytes (<=110KB)
    dim3 grid((M + 63) / 64);
    gemm_bias_wmma<<<grid, 256, sh, s>>>(A, lda, aoff, Aadd, W, bias,
                                         C, ldc, coff, M, K, relu);
}

extern "C" void kernel_launch(void* const* d_in, const int* in_sizes, int n_in,
                              void* d_out, int out_size, void* d_ws, size_t ws_size,
                              hipStream_t stream)
{
    (void)in_sizes; (void)n_in; (void)out_size; (void)ws_size;
    // Top-level dict in insertion order; 'params' expanded alphabetically.
    const float* atom_x          = (const float*)d_in[0];
    const float* atom_edge_attr  = (const float*)d_in[1];
    const float* motif_edge_attr = (const float*)d_in[2];
    const float* atom_enc_b = (const float*)d_in[3];
    const float* atom_enc_w = (const float*)d_in[4];
    const float* c1_b1 = (const float*)d_in[5];
    const float* c1_b2 = (const float*)d_in[6];
    const float* c1_eb = (const float*)d_in[7];
    const float* c1_ew = (const float*)d_in[8];
    const float* c1_w1 = (const float*)d_in[9];
    const float* c1_w2 = (const float*)d_in[10];
    const float* c2_b1 = (const float*)d_in[11];
    const float* c2_b2 = (const float*)d_in[12];
    const float* c2_eb = (const float*)d_in[13];
    const float* c2_ew = (const float*)d_in[14];
    const float* c2_w1 = (const float*)d_in[15];
    const float* c2_w2 = (const float*)d_in[16];
    const float* edge_enc_b = (const float*)d_in[17];
    const float* edge_enc_w = (const float*)d_in[18];
    const float* edge_nn_b  = (const float*)d_in[19];
    const float* edge_nn_w  = (const float*)d_in[20];
    const float* gin1_b1 = (const float*)d_in[21];
    const float* gin1_b2 = (const float*)d_in[22];
    const float* gin1_w1 = (const float*)d_in[23];
    const float* gin1_w2 = (const float*)d_in[24];
    const float* gin2_b1 = (const float*)d_in[25];
    const float* gin2_b2 = (const float*)d_in[26];
    const float* gin2_w1 = (const float*)d_in[27];
    const float* gin2_w2 = (const float*)d_in[28];
    const float* node_nn_b = (const float*)d_in[29];
    const float* node_nn_w = (const float*)d_in[30];
    const float* type_emb  = (const float*)d_in[31];
    const int* motif_type            = (const int*)d_in[32];
    const int* atom_edge_index       = (const int*)d_in[33];
    const int* motif_atom_edge_index = (const int*)d_in[34];
    const int* motif_edge_index      = (const int*)d_in[35];
    const int* motif_batch           = (const int*)d_in[36];
    const int* atom_ptr              = (const int*)d_in[37];

    // Outputs: [h (NM*H) | x_atom (NA*H) | motif_level (NG*H)]
    float* out_h  = (float*)d_out;
    float* out_x  = out_h + (size_t)N_MOTIFS * HID;
    float* out_ml = out_x + (size_t)N_ATOMS * HID;

    // Workspace layout (needs ~1.91 GB).
    float* ws = (float*)d_ws;
    size_t off = 0;
    float* e_h    = ws + off; off += (size_t)E_ATOM  * HID;
    float* bufA   = ws + off; off += (size_t)N_ATOMS * HID;   // msg / scratch
    float* bufB   = ws + off; off += (size_t)N_ATOMS * HID;   // x current
    float* bufC   = ws + off; off += (size_t)N_ATOMS * HID;   // hidden
    float* h_mot  = ws + off; off += (size_t)N_MOTIFS * HID;
    float* cat160 = ws + off; off += (size_t)N_MOTIFS * 160;
    float* h_e    = ws + off; off += (size_t)E_MOTIF * HID;
    float* cat288 = ws + off; off += (size_t)E_MOTIF * 288;
    float* e_edge = ws + off; off += (size_t)E_MOTIF * HID;
    float* msgM   = ws + off; off += (size_t)N_MOTIFS * HID;
    float* hm_tmp = ws + off; off += (size_t)N_MOTIFS * HID;
    float* hm_cur = ws + off; off += (size_t)N_MOTIFS * HID;

    const int* a_src = atom_edge_index;
    const int* a_dst = atom_edge_index + E_ATOM;
    const int* m_row = motif_atom_edge_index;            // motif ids
    const int* a_row = motif_atom_edge_index + N_ATOMS;  // atom ids
    const int* m_src = motif_edge_index;
    const int* m_dst = motif_edge_index + E_MOTIF;

    // 1) x_atom = relu(atom_x @ atom_enc_w + b)
    run_gemm(atom_x, 128, 0, nullptr, atom_enc_w, atom_enc_b,
             out_x, 128, 0, N_ATOMS, 128, 1, stream);
    // 2) e_h = relu(atom_edge_attr @ edge_enc_w + b)   (K=16, zero-padded)
    run_gemm(atom_edge_attr, 16, 0, nullptr, edge_enc_w, edge_enc_b,
             e_h, 128, 0, E_ATOM, 16, 1, stream);

    // 3) GINE layer 1 over atoms
    hipMemsetAsync(bufA, 0, (size_t)N_ATOMS * HID * 4, stream);
    gine_scatter<<<E_ATOM, HID, 0, stream>>>(out_x, e_h, a_src, a_dst, bufA);
    run_gemm(bufA, 128, 0, out_x, gin1_w1, gin1_b1, bufC, 128, 0, N_ATOMS, 128, 1, stream);
    run_gemm(bufC, 128, 0, nullptr, gin1_w2, gin1_b2, bufB, 128, 0, N_ATOMS, 128, 0, stream);

    // 4) GINE layer 2 over atoms
    hipMemsetAsync(bufA, 0, (size_t)N_ATOMS * HID * 4, stream);
    gine_scatter<<<E_ATOM, HID, 0, stream>>>(bufB, e_h, a_src, a_dst, bufA);
    run_gemm(bufA, 128, 0, bufB, gin2_w1, gin2_b1, bufC, 128, 0, N_ATOMS, 128, 1, stream);
    run_gemm(bufC, 128, 0, nullptr, gin2_w2, gin2_b2, bufB, 128, 0, N_ATOMS, 128, 0, stream);

    // 5) pool atoms -> motifs, then node_nn over [type_emb | h_motif]
    hipMemsetAsync(h_mot, 0, (size_t)N_MOTIFS * HID * 4, stream);
    pool_scatter<<<N_ATOMS, HID, 0, stream>>>(bufB, m_row, a_row, h_mot);
    build_cat160<<<N_MOTIFS, 160, 0, stream>>>(type_emb, motif_type, h_mot, cat160);
    run_gemm(cat160, 160, 0, nullptr, node_nn_w, node_nn_b,
             h_mot, 128, 0, N_MOTIFS, 160, 0, stream);

    // 6) motif edge embedding h_e
    motif_edge_prep<<<E_MOTIF, HID, 0, stream>>>(motif_edge_attr, m_src,
        motif_batch, atom_ptr, type_emb, out_x, cat288);
    run_gemm(motif_edge_attr, 20, 2, nullptr, edge_enc_w, edge_enc_b,
             cat288, 288, 32, E_MOTIF, 16, 1, stream);           // edge_emb slice
    run_gemm(cat288, 288, 0, nullptr, edge_nn_w, edge_nn_b,
             h_e, 128, 0, E_MOTIF, 288, 0, stream);

    // 7) GINE conv c1 over motifs
    run_gemm(h_e, 128, 0, nullptr, c1_ew, c1_eb, e_edge, 128, 0, E_MOTIF, 128, 0, stream);
    hipMemsetAsync(msgM, 0, (size_t)N_MOTIFS * HID * 4, stream);
    gine_scatter<<<E_MOTIF, HID, 0, stream>>>(h_mot, e_edge, m_src, m_dst, msgM);
    run_gemm(msgM, 128, 0, h_mot, c1_w1, c1_b1, hm_tmp, 128, 0, N_MOTIFS, 128, 1, stream);
    run_gemm(hm_tmp, 128, 0, nullptr, c1_w2, c1_b2, hm_cur, 128, 0, N_MOTIFS, 128, 0, stream);

    // 8) GINE conv c2 over motifs -> h (output 0)
    run_gemm(h_e, 128, 0, nullptr, c2_ew, c2_eb, e_edge, 128, 0, E_MOTIF, 128, 0, stream);
    hipMemsetAsync(msgM, 0, (size_t)N_MOTIFS * HID * 4, stream);
    gine_scatter<<<E_MOTIF, HID, 0, stream>>>(hm_cur, e_edge, m_src, m_dst, msgM);
    run_gemm(msgM, 128, 0, hm_cur, c2_w1, c2_b1, hm_tmp, 128, 0, N_MOTIFS, 128, 1, stream);
    run_gemm(hm_tmp, 128, 0, nullptr, c2_w2, c2_b2, out_h, 128, 0, N_MOTIFS, 128, 0, stream);

    // 9) motif_level = segment_sum(h, motif_batch)  (output 2)
    hipMemsetAsync(out_ml, 0, (size_t)N_GRAPHS * HID * 4, stream);
    graph_pool<<<N_MOTIFS, HID, 0, stream>>>(out_h, motif_batch, out_ml);
}
